// MoEDecoderLayer_2877628088863
// MI455X (gfx1250) — compile-verified, matrix operand order checked by reading, not run on hardware
//
#include <hip/hip_runtime.h>

typedef __bf16 bf16_t;
typedef bf16_t v16bf __attribute__((ext_vector_type(16)));
typedef bf16_t v8bf  __attribute__((ext_vector_type(8)));
typedef float  v8f   __attribute__((ext_vector_type(8)));

#define HDIM  2048
#define IDIM  1024
#define NEXP  32
#define TOPK  4
#define NTOK  4096
#define NPAIR 16384
#define SHI   2048

#define WMMA_BF16(A,B,C) \
  __builtin_amdgcn_wmma_f32_16x16x32_bf16(false,(A),false,(B),(short)0,(C),false,false)

__device__ __forceinline__ unsigned short f2bf(float f) {
  unsigned u = __float_as_uint(f);
  u += 0x7FFFu + ((u >> 16) & 1u);          // round-to-nearest-even
  return (unsigned short)(u >> 16);
}

__device__ __forceinline__ v8f zero8() {
  v8f z;
  for (int i = 0; i < 8; ++i) z[i] = 0.f;
  return z;
}

// Load one 16x32 bf16 fragment row-slice from LDS (stride 32 elems).
// ISA layout: lane&15 = M (or N), kbase = (lane>>4)*8; element e -> k = kbase + (e<8 ? e : e+8)
__device__ __forceinline__ v16bf frag(const unsigned short* lds, int row, int kb) {
  const v8bf lo = *reinterpret_cast<const v8bf*>(lds + row * 32 + kb);
  const v8bf hi = *reinterpret_cast<const v8bf*>(lds + row * 32 + kb + 16);
  return __builtin_shufflevector(lo, hi, 0,1,2,3,4,5,6,7,8,9,10,11,12,13,14,15);
}

// ---------------------------------------------------------------------------
// Kernel 1: RMSNorm + router (fp32 logits, softmax, top-4, renormalize)
// ---------------------------------------------------------------------------
__global__ __launch_bounds__(256) void k_norm_router(
    const float* __restrict__ x, const float* __restrict__ nw,
    const float* __restrict__ rw, unsigned short* __restrict__ flatn,
    float* __restrict__ topkw, int* __restrict__ topkid) {
  __shared__ float sx[HDIM];
  __shared__ float red[256];
  __shared__ float slog[NEXP];
  const int t = blockIdx.x, tid = threadIdx.x;
  const float* xr = x + (size_t)t * HDIM;

  float ss = 0.f;
  for (int i = tid; i < HDIM; i += 256) { float v = xr[i]; sx[i] = v; ss += v * v; }
  red[tid] = ss;
  __syncthreads();
  for (int s = 128; s > 0; s >>= 1) { if (tid < s) red[tid] += red[tid + s]; __syncthreads(); }
  const float rstd = rsqrtf(red[0] * (1.f / HDIM) + 1e-6f);

  for (int i = tid; i < HDIM; i += 256) {
    float h = sx[i] * rstd * nw[i];
    sx[i] = h;
    flatn[(size_t)t * HDIM + i] = f2bf(h);
  }
  __syncthreads();

  if (tid < NEXP) {
    const float* wr = rw + (size_t)tid * HDIM;
    float acc = 0.f;
    for (int i = 0; i < HDIM; ++i) acc += sx[i] * wr[i];
    slog[tid] = acc;
  }
  __syncthreads();

  if (tid == 0) {
    float mx = slog[0];
    for (int e = 1; e < NEXP; ++e) mx = fmaxf(mx, slog[e]);
    float sc[NEXP];
    for (int e = 0; e < NEXP; ++e) sc[e] = __expf(slog[e] - mx);
    unsigned mask = 0; float wsum = 0.f;
    int   id[TOPK]; float wv[TOPK];
    for (int k = 0; k < TOPK; ++k) {
      int bi = 0; float bv = -1.f;
      for (int e = 0; e < NEXP; ++e)
        if (!((mask >> e) & 1u) && sc[e] > bv) { bv = sc[e]; bi = e; }
      mask |= 1u << bi; id[k] = bi; wv[k] = bv; wsum += bv;
    }
    const float inv = 1.f / wsum;
    for (int k = 0; k < TOPK; ++k) {
      topkw[t * TOPK + k]  = wv[k] * inv;
      topkid[t * TOPK + k] = id[k];
    }
  }
}

// ---------------------------------------------------------------------------
// Dispatch: zero meta, count, scan, scatter
// ---------------------------------------------------------------------------
__global__ void k_init(int* meta) { if (threadIdx.x < 96) meta[threadIdx.x] = 0; }

__global__ void k_count(const int* __restrict__ ids, int* __restrict__ cnt) {
  int p = blockIdx.x * 256 + threadIdx.x;
  if (p < NPAIR) atomicAdd(&cnt[ids[p]], 1);
}

__global__ void k_scan(const int* __restrict__ cnt, int* __restrict__ seg) {
  if (threadIdx.x == 0) {
    int s = 0;
    for (int e = 0; e < NEXP; ++e) { seg[e] = s; s += cnt[e]; }
  }
}

__global__ void k_scatter(const int* __restrict__ ids, const int* __restrict__ seg,
                          int* __restrict__ cursor, int* __restrict__ rowPair) {
  int p = blockIdx.x * 256 + threadIdx.x;
  if (p < NPAIR) {
    int e = ids[p];
    int pos = seg[e] + atomicAdd(&cursor[e], 1);
    rowPair[pos] = p;
  }
}

// ---------------------------------------------------------------------------
// GEMM1: act = silu(xs@w1_g) * (xs@w1_u)   per expert, gathered rows
// grid: x = colTile (I/128 = 8), y = expert (32); row-blocks of 64 looped
// ---------------------------------------------------------------------------
__global__ __launch_bounds__(256) void k_gemm1(
    const unsigned short* __restrict__ flatn, const float* __restrict__ w1,
    const int* __restrict__ rowPair, const int* __restrict__ seg,
    const int* __restrict__ cntArr, unsigned short* __restrict__ act) {
  __shared__ __align__(16) unsigned short ldsA[64 * 32];
  __shared__ __align__(16) unsigned short ldsBg[128 * 32];
  __shared__ __align__(16) unsigned short ldsBu[128 * 32];
  const int ct = blockIdx.x, e = blockIdx.y;
  const int cnt  = cntArr[e];
  const int segS = seg[e];
  const int tid = threadIdx.x, lane = tid & 31;
  const int wave = tid >> 5;
  const int wm = (wave >> 2) * 32, wn = (wave & 3) * 32;
  const int kb = (lane >> 4) * 8, ml = lane & 15;
  const float* wbase = w1 + (size_t)e * HDIM * (2 * IDIM);

  for (int rowBase = 0; rowBase < cnt; rowBase += 64) {
    v8f aG[2][2], aU[2][2];
    for (int i = 0; i < 2; ++i)
      for (int j = 0; j < 2; ++j) { aG[i][j] = zero8(); aU[i][j] = zero8(); }

    for (int k0 = 0; k0 < HDIM; k0 += 32) {
      { // A tile (gathered bf16 rows)
        int r = tid >> 2, c = tid & 3;
        int gr = rowBase + r;
        int tok = (gr < cnt) ? (rowPair[segS + gr] >> 2) : 0;
        ((uint4*)ldsA)[tid] = ((const uint4*)(flatn + (size_t)tok * HDIM + k0))[c];
      }
      { // B tiles g/u (fp32 -> bf16, transpose-store as [n][k])
        int tn = (tid & 31) * 4, tk = tid >> 5;
        for (int kk = tk; kk < 32; kk += 8) {
          const float* gsrc = wbase + (size_t)(k0 + kk) * (2 * IDIM) + ct * 128 + tn;
          if (k0 + 32 < HDIM) __builtin_prefetch(gsrc + 32 * (2 * IDIM), 0, 0);
          float4 gv = *(const float4*)gsrc;
          float4 uv = *(const float4*)(gsrc + IDIM);
          ldsBg[(tn + 0) * 32 + kk] = f2bf(gv.x); ldsBg[(tn + 1) * 32 + kk] = f2bf(gv.y);
          ldsBg[(tn + 2) * 32 + kk] = f2bf(gv.z); ldsBg[(tn + 3) * 32 + kk] = f2bf(gv.w);
          ldsBu[(tn + 0) * 32 + kk] = f2bf(uv.x); ldsBu[(tn + 1) * 32 + kk] = f2bf(uv.y);
          ldsBu[(tn + 2) * 32 + kk] = f2bf(uv.z); ldsBu[(tn + 3) * 32 + kk] = f2bf(uv.w);
        }
      }
      __syncthreads();
      v16bf a0  = frag(ldsA,  wm + ml,      kb);
      v16bf a1  = frag(ldsA,  wm + 16 + ml, kb);
      v16bf bg0 = frag(ldsBg, wn + ml,      kb);
      v16bf bg1 = frag(ldsBg, wn + 16 + ml, kb);
      v16bf bu0 = frag(ldsBu, wn + ml,      kb);
      v16bf bu1 = frag(ldsBu, wn + 16 + ml, kb);
      aG[0][0] = WMMA_BF16(a0, bg0, aG[0][0]); aG[0][1] = WMMA_BF16(a0, bg1, aG[0][1]);
      aG[1][0] = WMMA_BF16(a1, bg0, aG[1][0]); aG[1][1] = WMMA_BF16(a1, bg1, aG[1][1]);
      aU[0][0] = WMMA_BF16(a0, bu0, aU[0][0]); aU[0][1] = WMMA_BF16(a0, bu1, aU[0][1]);
      aU[1][0] = WMMA_BF16(a1, bu0, aU[1][0]); aU[1][1] = WMMA_BF16(a1, bu1, aU[1][1]);
      __syncthreads();
    }
    // epilogue: SwiGLU, store bf16 act at sorted position
    for (int i = 0; i < 2; ++i)
      for (int j = 0; j < 2; ++j)
        for (int r = 0; r < 8; ++r) {
          int mloc = r + ((lane >> 4) << 3);
          int gr = rowBase + wm + i * 16 + mloc;
          if (gr < cnt) {
            int n = ct * 128 + wn + j * 16 + (lane & 15);
            float g = aG[i][j][r], u = aU[i][j][r];
            float s = g * (1.f / (1.f + __expf(-g))) * u;
            act[(size_t)(segS + gr) * IDIM + n] = f2bf(s);
          }
        }
  }
}

// ---------------------------------------------------------------------------
// GEMM2: ys[pair] = act @ w2[e]    grid: x = H/128 = 16, y = expert
// ---------------------------------------------------------------------------
__global__ __launch_bounds__(256) void k_gemm2(
    const unsigned short* __restrict__ act, const float* __restrict__ w2,
    const int* __restrict__ rowPair, const int* __restrict__ seg,
    const int* __restrict__ cntArr, float* __restrict__ ys) {
  __shared__ __align__(16) unsigned short ldsA[64 * 32];
  __shared__ __align__(16) unsigned short ldsB[128 * 32];
  const int ct = blockIdx.x, e = blockIdx.y;
  const int cnt  = cntArr[e];
  const int segS = seg[e];
  const int tid = threadIdx.x, lane = tid & 31;
  const int wave = tid >> 5;
  const int wm = (wave >> 2) * 32, wn = (wave & 3) * 32;
  const int kb = (lane >> 4) * 8, ml = lane & 15;
  const float* wbase = w2 + (size_t)e * IDIM * HDIM;

  for (int rowBase = 0; rowBase < cnt; rowBase += 64) {
    v8f acc[2][2];
    for (int i = 0; i < 2; ++i)
      for (int j = 0; j < 2; ++j) acc[i][j] = zero8();

    for (int k0 = 0; k0 < IDIM; k0 += 32) {
      {
        int r = tid >> 2, c = tid & 3;
        int gr = rowBase + r;
        int row = segS + ((gr < cnt) ? gr : 0);
        ((uint4*)ldsA)[tid] = ((const uint4*)(act + (size_t)row * IDIM + k0))[c];
      }
      {
        int tn = (tid & 31) * 4, tk = tid >> 5;
        for (int kk = tk; kk < 32; kk += 8) {
          const float* src = wbase + (size_t)(k0 + kk) * HDIM + ct * 128 + tn;
          if (k0 + 32 < IDIM) __builtin_prefetch(src + 32 * HDIM, 0, 0);
          float4 v = *(const float4*)src;
          ldsB[(tn + 0) * 32 + kk] = f2bf(v.x); ldsB[(tn + 1) * 32 + kk] = f2bf(v.y);
          ldsB[(tn + 2) * 32 + kk] = f2bf(v.z); ldsB[(tn + 3) * 32 + kk] = f2bf(v.w);
        }
      }
      __syncthreads();
      v16bf a0 = frag(ldsA, wm + ml,      kb);
      v16bf a1 = frag(ldsA, wm + 16 + ml, kb);
      v16bf b0 = frag(ldsB, wn + ml,      kb);
      v16bf b1 = frag(ldsB, wn + 16 + ml, kb);
      acc[0][0] = WMMA_BF16(a0, b0, acc[0][0]); acc[0][1] = WMMA_BF16(a0, b1, acc[0][1]);
      acc[1][0] = WMMA_BF16(a1, b0, acc[1][0]); acc[1][1] = WMMA_BF16(a1, b1, acc[1][1]);
      __syncthreads();
    }
    for (int i = 0; i < 2; ++i)
      for (int j = 0; j < 2; ++j)
        for (int r = 0; r < 8; ++r) {
          int mloc = r + ((lane >> 4) << 3);
          int gr = rowBase + wm + i * 16 + mloc;
          if (gr < cnt) {
            int pair = rowPair[segS + gr];
            int n = ct * 128 + wn + j * 16 + (lane & 15);
            ys[(size_t)pair * HDIM + n] = acc[i][j][r];
          }
        }
  }
}

// ---------------------------------------------------------------------------
// Shared expert up: sact = silu(flatn@sg) * (flatn@su)
// grid: x = SHI/128 = 16, y = T/64 = 64
// ---------------------------------------------------------------------------
__global__ __launch_bounds__(256) void k_shared1(
    const unsigned short* __restrict__ flatn, const float* __restrict__ sg,
    const float* __restrict__ su, unsigned short* __restrict__ sact) {
  __shared__ __align__(16) unsigned short ldsA[64 * 32];
  __shared__ __align__(16) unsigned short ldsBg[128 * 32];
  __shared__ __align__(16) unsigned short ldsBu[128 * 32];
  const int ct = blockIdx.x, rowBase = blockIdx.y * 64;
  const int tid = threadIdx.x, lane = tid & 31;
  const int wave = tid >> 5;
  const int wm = (wave >> 2) * 32, wn = (wave & 3) * 32;
  const int kb = (lane >> 4) * 8, ml = lane & 15;

  v8f aG[2][2], aU[2][2];
  for (int i = 0; i < 2; ++i)
    for (int j = 0; j < 2; ++j) { aG[i][j] = zero8(); aU[i][j] = zero8(); }

  for (int k0 = 0; k0 < HDIM; k0 += 32) {
    {
      int r = tid >> 2, c = tid & 3;
      int tok = rowBase + r;
      ((uint4*)ldsA)[tid] = ((const uint4*)(flatn + (size_t)tok * HDIM + k0))[c];
    }
    {
      int tn = (tid & 31) * 4, tk = tid >> 5;
      for (int kk = tk; kk < 32; kk += 8) {
        size_t o = (size_t)(k0 + kk) * SHI + ct * 128 + tn;
        if (k0 + 32 < HDIM) __builtin_prefetch(sg + o + 32 * SHI, 0, 0);
        float4 gv = *(const float4*)(sg + o);
        float4 uv = *(const float4*)(su + o);
        ldsBg[(tn + 0) * 32 + kk] = f2bf(gv.x); ldsBg[(tn + 1) * 32 + kk] = f2bf(gv.y);
        ldsBg[(tn + 2) * 32 + kk] = f2bf(gv.z); ldsBg[(tn + 3) * 32 + kk] = f2bf(gv.w);
        ldsBu[(tn + 0) * 32 + kk] = f2bf(uv.x); ldsBu[(tn + 1) * 32 + kk] = f2bf(uv.y);
        ldsBu[(tn + 2) * 32 + kk] = f2bf(uv.z); ldsBu[(tn + 3) * 32 + kk] = f2bf(uv.w);
      }
    }
    __syncthreads();
    v16bf a0  = frag(ldsA,  wm + ml,      kb);
    v16bf a1  = frag(ldsA,  wm + 16 + ml, kb);
    v16bf bg0 = frag(ldsBg, wn + ml,      kb);
    v16bf bg1 = frag(ldsBg, wn + 16 + ml, kb);
    v16bf bu0 = frag(ldsBu, wn + ml,      kb);
    v16bf bu1 = frag(ldsBu, wn + 16 + ml, kb);
    aG[0][0] = WMMA_BF16(a0, bg0, aG[0][0]); aG[0][1] = WMMA_BF16(a0, bg1, aG[0][1]);
    aG[1][0] = WMMA_BF16(a1, bg0, aG[1][0]); aG[1][1] = WMMA_BF16(a1, bg1, aG[1][1]);
    aU[0][0] = WMMA_BF16(a0, bu0, aU[0][0]); aU[0][1] = WMMA_BF16(a0, bu1, aU[0][1]);
    aU[1][0] = WMMA_BF16(a1, bu0, aU[1][0]); aU[1][1] = WMMA_BF16(a1, bu1, aU[1][1]);
    __syncthreads();
  }
  for (int i = 0; i < 2; ++i)
    for (int j = 0; j < 2; ++j)
      for (int r = 0; r < 8; ++r) {
        int mloc = r + ((lane >> 4) << 3);
        int t = rowBase + wm + i * 16 + mloc;
        int n = ct * 128 + wn + j * 16 + (lane & 15);
        float g = aG[i][j][r], u = aU[i][j][r];
        float s = g * (1.f / (1.f + __expf(-g))) * u;
        sact[(size_t)t * SHI + n] = f2bf(s);
      }
}

// ---------------------------------------------------------------------------
// Shared expert down + residual: out = x + sact @ sd
// grid: x = H/128 = 16, y = T/64 = 64
// ---------------------------------------------------------------------------
__global__ __launch_bounds__(256) void k_shared2(
    const unsigned short* __restrict__ sact, const float* __restrict__ sd,
    const float* __restrict__ xres, float* __restrict__ out) {
  __shared__ __align__(16) unsigned short ldsA[64 * 32];
  __shared__ __align__(16) unsigned short ldsB[128 * 32];
  const int ct = blockIdx.x, rowBase = blockIdx.y * 64;
  const int tid = threadIdx.x, lane = tid & 31;
  const int wave = tid >> 5;
  const int wm = (wave >> 2) * 32, wn = (wave & 3) * 32;
  const int kb = (lane >> 4) * 8, ml = lane & 15;

  v8f acc[2][2];
  for (int i = 0; i < 2; ++i)
    for (int j = 0; j < 2; ++j) acc[i][j] = zero8();

  for (int k0 = 0; k0 < SHI; k0 += 32) {
    {
      int r = tid >> 2, c = tid & 3;
      int tok = rowBase + r;
      ((uint4*)ldsA)[tid] = ((const uint4*)(sact + (size_t)tok * SHI + k0))[c];
    }
    {
      int tn = (tid & 31) * 4, tk = tid >> 5;
      for (int kk = tk; kk < 32; kk += 8) {
        const float* src = sd + (size_t)(k0 + kk) * HDIM + ct * 128 + tn;
        if (k0 + 32 < SHI) __builtin_prefetch(src + 32 * HDIM, 0, 0);
        float4 v = *(const float4*)src;
        ldsB[(tn + 0) * 32 + kk] = f2bf(v.x); ldsB[(tn + 1) * 32 + kk] = f2bf(v.y);
        ldsB[(tn + 2) * 32 + kk] = f2bf(v.z); ldsB[(tn + 3) * 32 + kk] = f2bf(v.w);
      }
    }
    __syncthreads();
    v16bf a0 = frag(ldsA, wm + ml,      kb);
    v16bf a1 = frag(ldsA, wm + 16 + ml, kb);
    v16bf b0 = frag(ldsB, wn + ml,      kb);
    v16bf b1 = frag(ldsB, wn + 16 + ml, kb);
    acc[0][0] = WMMA_BF16(a0, b0, acc[0][0]); acc[0][1] = WMMA_BF16(a0, b1, acc[0][1]);
    acc[1][0] = WMMA_BF16(a1, b0, acc[1][0]); acc[1][1] = WMMA_BF16(a1, b1, acc[1][1]);
    __syncthreads();
  }
  for (int i = 0; i < 2; ++i)
    for (int j = 0; j < 2; ++j)
      for (int r = 0; r < 8; ++r) {
        int mloc = r + ((lane >> 4) << 3);
        int t = rowBase + wm + i * 16 + mloc;
        int n = ct * 128 + wn + j * 16 + (lane & 15);
        size_t o = (size_t)t * HDIM + n;
        out[o] = xres[o] + acc[i][j][r];
      }
}

// ---------------------------------------------------------------------------
// Combine: out += sum_k w[t,k] * ys[t*4+k]
// ---------------------------------------------------------------------------
__global__ __launch_bounds__(256) void k_combine(
    const float* __restrict__ ys, const float* __restrict__ topkw,
    float* __restrict__ out) {
  const int t = blockIdx.x, tid = threadIdx.x;
  const float w0 = topkw[t * 4 + 0], w1 = topkw[t * 4 + 1];
  const float w2 = topkw[t * 4 + 2], w3 = topkw[t * 4 + 3];
  const float* y0 = ys + (size_t)(t * 4 + 0) * HDIM;
  const float* y1 = ys + (size_t)(t * 4 + 1) * HDIM;
  const float* y2 = ys + (size_t)(t * 4 + 2) * HDIM;
  const float* y3 = ys + (size_t)(t * 4 + 3) * HDIM;
  float* o = out + (size_t)t * HDIM;
  for (int h = tid; h < HDIM; h += 256)
    o[h] += w0 * y0[h] + w1 * y1[h] + w2 * y2[h] + w3 * y3[h];
}

// ---------------------------------------------------------------------------
extern "C" void kernel_launch(void* const* d_in, const int* in_sizes, int n_in,
                              void* d_out, int out_size, void* d_ws, size_t ws_size,
                              hipStream_t stream) {
  const float* x   = (const float*)d_in[0];
  const float* nw  = (const float*)d_in[1];
  const float* rw  = (const float*)d_in[2];
  const float* w1  = (const float*)d_in[3];
  const float* w2  = (const float*)d_in[4];
  const float* sg  = (const float*)d_in[5];
  const float* su  = (const float*)d_in[6];
  const float* sd  = (const float*)d_in[7];
  float* out = (float*)d_out;

  char* ws = (char*)d_ws;
  size_t off = 0;
  auto take = [&](size_t bytes) {
    void* p = ws + off;
    off = (off + bytes + 255) & ~(size_t)255;
    return p;
  };
  unsigned short* flatn  = (unsigned short*)take((size_t)NTOK * HDIM * 2);   // 16 MB
  unsigned short* act    = (unsigned short*)take((size_t)NPAIR * IDIM * 2);  // 32 MB
  unsigned short* sact   = (unsigned short*)take((size_t)NTOK * SHI * 2);    // 16 MB
  float*          ys     = (float*)take((size_t)NPAIR * HDIM * 4);           // 128 MB
  float*          topkw  = (float*)take((size_t)NTOK * TOPK * 4);
  int*            topkid = (int*)take((size_t)NTOK * TOPK * 4);
  int*            rowPair= (int*)take((size_t)NPAIR * 4);
  int*            meta   = (int*)take(1024);
  int* cnt = meta, *seg = meta + 32, *cursor = meta + 64;

  k_init<<<1, 128, 0, stream>>>(meta);
  k_norm_router<<<NTOK, 256, 0, stream>>>(x, nw, rw, flatn, topkw, topkid);
  k_count<<<NPAIR / 256, 256, 0, stream>>>(topkid, cnt);
  k_scan<<<1, 32, 0, stream>>>(cnt, seg);
  k_scatter<<<NPAIR / 256, 256, 0, stream>>>(topkid, seg, cursor, rowPair);
  k_gemm1<<<dim3(IDIM / 128, NEXP), 256, 0, stream>>>(flatn, w1, rowPair, seg, cnt, act);
  k_gemm2<<<dim3(HDIM / 128, NEXP), 256, 0, stream>>>(act, w2, rowPair, seg, cnt, ys);
  k_shared1<<<dim3(SHI / 128, NTOK / 64), 256, 0, stream>>>(flatn, sg, su, sact);
  k_shared2<<<dim3(HDIM / 128, NTOK / 64), 256, 0, stream>>>(sact, sd, x, out);
  k_combine<<<NTOK, 256, 0, stream>>>(ys, topkw, out);
}
